// SchNetEncoder_26079041421823
// MI455X (gfx1250) — matrix operand order repaired and checked by reading, not compiled, not run on hardware
//
#include <hip/hip_runtime.h>
#include <math.h>

// ---------------------------------------------------------------------------
// SchNet encoder for MI455X (gfx1250, wave32, WMMA).
// ---------------------------------------------------------------------------

typedef __attribute__((ext_vector_type(16))) _Float16 v16h;
typedef __attribute__((ext_vector_type(8)))  _Float16 v8h;
typedef __attribute__((ext_vector_type(8)))  float    v8f;

#define NNODES 100000
#define NEDGES 320000
#define HDIM   256
#define GDIM   50
#define GPAD   64       // pad Gaussian dim to a multiple of K=32
#define LAYERS 6
#define PDIM   256
#define PI_F   3.14159265358979f
#define LN2_F  0.69314718055994531f

__device__ __forceinline__ float sspf(float x) {
    float sp = (x > 20.f) ? x : log1pf(__expf(x));
    return sp - LN2_F;
}

// ---------------------------------------------------------------------------
// Templated WMMA GEMM: C[M,Nc] = A[M,K] @ B[K,Nc]  (B pre-converted f16)
// Block tile 128x32, 8 wave32 waves; each wave computes a 16x32 strip
// (2 accumulators, A fragment reused). K staged through LDS in steps of 32
// with double-buffered (ping-pong) staging: one barrier per iteration, next
// tile's global loads overlap the current tile's WMMAs.
//
// LDS A is stored in natural K order (AsP[row][k]); lane half h reads k = 16h..16h+15
// so a lane's 16-half fragment is 32 contiguous bytes (2x ds_load_b128).
// LDS B is stored transposed (BsT[col][k], natural k order) for the same reason.
// ---------------------------------------------------------------------------
enum { EPI_F16 = 0, EPI_SSP_F16 = 1, EPI_SCATTER = 2, EPI_RESID = 3, EPI_F32 = 4 };

template <int EPI>
__device__ __forceinline__
void epi_store(int row, int col, float v, int Nc,
               const float* __restrict__ bias,
               _Float16* __restrict__ out16, float* __restrict__ out32,
               const float* __restrict__ edgeC,
               const int* __restrict__ esrc, const int* __restrict__ edst,
               const _Float16* __restrict__ x1, float* __restrict__ agg) {
    if (bias) v += bias[col];
    if (EPI == EPI_F16) {
        out16[(size_t)row * Nc + col] = (_Float16)v;
    } else if (EPI == EPI_SSP_F16) {
        out16[(size_t)row * Nc + col] = (_Float16)sspf(v);
    } else if (EPI == EPI_SCATTER) {
        // row == edge id: Wf = (acc+b)*C[e]; msg = x1[src]*Wf; agg[dst] += msg
        v *= edgeC[row];
        const int s = esrc[row];
        const int d = edst[row];
        const float msg = (float)x1[(size_t)s * HDIM + col] * v;
        __hip_atomic_fetch_add(&agg[(size_t)d * HDIM + col], msg,
                               __ATOMIC_RELAXED, __HIP_MEMORY_SCOPE_AGENT);
    } else if (EPI == EPI_RESID) {
        out32[(size_t)row * Nc + col] += v;
    } else { // EPI_F32
        out32[(size_t)row * Nc + col] = v;
    }
}

template <int EPI, typename TA>
__global__ __launch_bounds__(256)
void wmma_gemm_kernel(const TA* __restrict__ A, int lda, int M,
                      const _Float16* __restrict__ B, int Nc, int K,
                      const float* __restrict__ bias,
                      _Float16* __restrict__ out16,
                      float* __restrict__ out32,
                      const float* __restrict__ edgeC,
                      const int* __restrict__ esrc,
                      const int* __restrict__ edst,
                      const _Float16* __restrict__ x1,
                      float* __restrict__ agg)
{
    // stride 40 halfs (80 B): rows 16B-aligned, 20*row mod 64 distinct for 16 rows
    __shared__ alignas(16) _Float16 AsP[2][128][40];
    __shared__ alignas(16) _Float16 BsT[2][32][40];

    const int tid  = threadIdx.x;
    const int lane = tid & 31;
    const int wave = tid >> 5;
    const int lg   = lane >> 4;
    const int ln16 = lane & 15;
    const int wm   = wave * 16;            // wave strip row within 128-row tile
    const int row0 = blockIdx.x * 128;
    const int col0 = blockIdx.y * 32;

    // A staging: thread handles one row, 16 halfs (two 8-chunks, natural k order)
    const int sArow = tid >> 1;            // 0..127
    const int sAk   = (tid & 1) * 16;      // k chunk base: 0 or 16
    const int p0    = sAk;                 // natural pos of chunk k=sAk
    const int p1    = sAk + 8;             // natural pos of chunk k=sAk+8
    // B staging: thread loads 4 halfs of one k-row, writes transposed
    const int sBk   = tid >> 3;            // 0..31
    const int sBc0  = (tid & 7) * 4;

    const int garow = row0 + sArow;

    auto stage = [&](int buf, int kst) {
        _Float16 hv[16];
        if (garow < M) {
            const TA* ap = A + (size_t)garow * lda + kst + sAk;
#pragma unroll
            for (int j = 0; j < 16; ++j) hv[j] = (_Float16)ap[j];
            if (kst + 32 < K) __builtin_prefetch(ap + 32, 0, 1);
        } else {
#pragma unroll
            for (int j = 0; j < 16; ++j) hv[j] = (_Float16)0.f;
        }
        v8h h0 = {hv[0], hv[1], hv[2],  hv[3],  hv[4],  hv[5],  hv[6],  hv[7]};
        v8h h1 = {hv[8], hv[9], hv[10], hv[11], hv[12], hv[13], hv[14], hv[15]};
        *(v8h*)&AsP[buf][sArow][p0] = h0;
        *(v8h*)&AsP[buf][sArow][p1] = h1;
        const _Float16* bp = B + (size_t)(kst + sBk) * Nc + col0 + sBc0;
#pragma unroll
        for (int j = 0; j < 4; ++j) BsT[buf][sBc0 + j][sBk] = bp[j];
    };

    v8f acc0 = {};
    v8f acc1 = {};

    stage(0, 0);
    int pb = 0;
    for (int k0 = 0; k0 < K; k0 += 32) {
        __syncthreads();   // staged tile pb visible; prior reads of pb^1 done
        if (k0 + 32 < K) stage(pb ^ 1, k0 + 32);

        // ---- fragments: 32 contiguous bytes per lane ----
        const v8h a0 = *(const v8h*)&AsP[pb][wm + ln16][lg * 16];
        const v8h a1 = *(const v8h*)&AsP[pb][wm + ln16][lg * 16 + 8];
        const v16h af = __builtin_shufflevector(a0, a1,
            0,1,2,3,4,5,6,7,8,9,10,11,12,13,14,15);
        const v8h b00 = *(const v8h*)&BsT[pb][ln16][lg * 16];
        const v8h b01 = *(const v8h*)&BsT[pb][ln16][lg * 16 + 8];
        const v16h bf0 = __builtin_shufflevector(b00, b01,
            0,1,2,3,4,5,6,7,8,9,10,11,12,13,14,15);
        const v8h b10 = *(const v8h*)&BsT[pb][16 + ln16][lg * 16];
        const v8h b11 = *(const v8h*)&BsT[pb][16 + ln16][lg * 16 + 8];
        const v16h bf1 = __builtin_shufflevector(b10, b11,
            0,1,2,3,4,5,6,7,8,9,10,11,12,13,14,15);

        acc0 = __builtin_amdgcn_wmma_f32_16x16x32_f16(
            false, af, false, bf0, (short)0, acc0, false, false);
        acc1 = __builtin_amdgcn_wmma_f32_16x16x32_f16(
            false, af, false, bf1, (short)0, acc1, false, false);
        pb ^= 1;
    }

    // ---- epilogue: VGPR r holds M=r (+8 for lanes 16..31), N=lane%16 ----
    const int c0 = col0 + ln16;
    const int c1 = col0 + 16 + ln16;
#pragma unroll
    for (int r = 0; r < 8; ++r) {
        const int row = row0 + wm + r + 8 * lg;
        if (row >= M) continue;
        epi_store<EPI>(row, c0, acc0[r], Nc, bias, out16, out32,
                       edgeC, esrc, edst, x1, agg);
        epi_store<EPI>(row, c1, acc1[r], Nc, bias, out16, out32,
                       edgeC, esrc, edst, x1, agg);
    }
}

// ---------------------------------------------------------------------------
// Elementwise helpers
// ---------------------------------------------------------------------------
__global__ void f32_to_f16_kernel(const float* __restrict__ in,
                                  _Float16* __restrict__ out, int n) {
    int i = blockIdx.x * blockDim.x + threadIdx.x;
    if (i < n) out[i] = (_Float16)in[i];
}

__global__ void pad_w1_kernel(const float* __restrict__ in,
                              _Float16* __restrict__ out) {
    int i = blockIdx.x * blockDim.x + threadIdx.x; // L*GPAD*H
    if (i >= LAYERS * GPAD * HDIM) return;
    const int c = i % HDIM;
    const int g = (i / HDIM) % GPAD;
    const int l = i / (HDIM * GPAD);
    out[i] = (g < GDIM) ? (_Float16)in[((size_t)l * GDIM + g) * HDIM + c]
                        : (_Float16)0.f;
}

__global__ void fill_zero_kernel(float* __restrict__ p, int n) {
    int i = blockIdx.x * blockDim.x + threadIdx.x;
    if (i < n) p[i] = 0.f;
}

__global__ void embed_kernel(const int* __restrict__ z,
                             const float* __restrict__ emb,
                             float* __restrict__ h, int n) {
    int i = blockIdx.x * blockDim.x + threadIdx.x; // N*H
    if (i >= n) return;
    const int node = i / HDIM, c = i % HDIM;
    h[i] = emb[(size_t)z[node] * HDIM + c];
}

// distances, cosine cutoff, Gaussian smearing (f16, zero-padded to GPAD)
__global__ void edge_geom_kernel(const float* __restrict__ pos,
                                 const int* __restrict__ ei,
                                 _Float16* __restrict__ ea,
                                 float* __restrict__ C) {
    int e = blockIdx.x * blockDim.x + threadIdx.x;
    if (e >= NEDGES) return;
    const int s = ei[e], d = ei[NEDGES + e];
    const float dx = pos[s * 3 + 0] - pos[d * 3 + 0];
    const float dy = pos[s * 3 + 1] - pos[d * 3 + 1];
    const float dz = pos[s * 3 + 2] - pos[d * 3 + 2];
    const float dist = sqrtf(dx * dx + dy * dy + dz * dz + 1e-12f);
    C[e] = 0.5f * (__cosf(dist * PI_F / 5.0f) + 1.0f);
    const float step  = 5.0f / 49.0f;
    const float coeff = -0.5f / (step * step);
#pragma unroll 8
    for (int g = 0; g < GPAD; ++g) {
        float v = 0.f;
        if (g < GDIM) {
            const float t = dist - (float)g * step;
            v = __expf(coeff * t * t);
        }
        ea[(size_t)e * GPAD + g] = (_Float16)v;
    }
}

// fused LayerNorm + SiLU, one 256-thread block per node row (P == 256)
__global__ __launch_bounds__(256)
void ln_silu_kernel(float* __restrict__ p,
                    const float* __restrict__ g,
                    const float* __restrict__ b) {
    __shared__ float red[256];
    const int row = blockIdx.x;
    const int c = threadIdx.x;
    const float v = p[(size_t)row * PDIM + c];
    red[c] = v;
    __syncthreads();
    for (int s = 128; s > 0; s >>= 1) {
        if (c < s) red[c] += red[c + s];
        __syncthreads();
    }
    const float mu = red[0] / (float)PDIM;
    __syncthreads();
    const float dlt = v - mu;
    red[c] = dlt * dlt;
    __syncthreads();
    for (int s = 128; s > 0; s >>= 1) {
        if (c < s) red[c] += red[c + s];
        __syncthreads();
    }
    const float var = red[0] / (float)PDIM;
    float y = dlt * rsqrtf(var + 1e-5f) * g[c] + b[c];
    y = y / (1.0f + __expf(-y)); // SiLU
    p[(size_t)row * PDIM + c] = y;
}

// ---------------------------------------------------------------------------
// launch
// ---------------------------------------------------------------------------
extern "C" void kernel_launch(void* const* d_in, const int* in_sizes, int n_in,
                              void* d_out, int out_size, void* d_ws, size_t ws_size,
                              hipStream_t stream) {
    (void)in_sizes; (void)n_in; (void)out_size; (void)ws_size;

    const int*   z      = (const int*)d_in[0];
    const float* pos    = (const float*)d_in[1];
    const int*   batch  = (const int*)d_in[2];
    const int*   ei     = (const int*)d_in[3];
    const float* emb    = (const float*)d_in[4];
    const float* mlp_w1 = (const float*)d_in[5];
    const float* mlp_b1 = (const float*)d_in[6];
    const float* mlp_w2 = (const float*)d_in[7];
    const float* mlp_b2 = (const float*)d_in[8];
    const float* lin1_w = (const float*)d_in[9];
    const float* lin2_w = (const float*)d_in[10];
    const float* lin2_b = (const float*)d_in[11];
    const float* lin_w  = (const float*)d_in[12];
    const float* lin_b  = (const float*)d_in[13];
    const float* proj_w = (const float*)d_in[14];
    const float* proj_b = (const float*)d_in[15];
    const float* ln_g   = (const float*)d_in[16];
    const float* ln_b   = (const float*)d_in[17];

    // ---- workspace carve-up (256B aligned) ----
    char* w = (char*)d_ws;
    auto carve = [&](size_t bytes) -> char* {
        char* p = w;
        w += (bytes + 255) & ~(size_t)255;
        return p;
    };
    const size_t HH = (size_t)HDIM * HDIM;
    _Float16* w1h   = (_Float16*)carve(sizeof(_Float16) * LAYERS * GPAD * HDIM);
    _Float16* w2h   = (_Float16*)carve(sizeof(_Float16) * LAYERS * HH);
    _Float16* l1h   = (_Float16*)carve(sizeof(_Float16) * LAYERS * HH);
    _Float16* l2h   = (_Float16*)carve(sizeof(_Float16) * LAYERS * HH);
    _Float16* lwh   = (_Float16*)carve(sizeof(_Float16) * LAYERS * HH);
    _Float16* pjh   = (_Float16*)carve(sizeof(_Float16) * (size_t)HDIM * PDIM);
    _Float16* ea    = (_Float16*)carve(sizeof(_Float16) * (size_t)NEDGES * GPAD);
    float*    Cbuf  = (float*)   carve(sizeof(float)    * (size_t)NEDGES);
    _Float16* tbuf  = (_Float16*)carve(sizeof(_Float16) * (size_t)NEDGES * HDIM);
    _Float16* x1h   = (_Float16*)carve(sizeof(_Float16) * (size_t)NNODES * HDIM);
    float*    aggb  = (float*)   carve(sizeof(float)    * (size_t)NNODES * HDIM);
    _Float16* sbuf  = (_Float16*)carve(sizeof(_Float16) * (size_t)NNODES * HDIM);
    float*    hbuf  = (float*)   carve(sizeof(float)    * (size_t)NNODES * HDIM);

    const int NH = NNODES * HDIM;

    // ---- weight conversion (f32 -> f16, w1 padded 50 -> 64) ----
    {
        const int nW1 = LAYERS * GPAD * HDIM;
        pad_w1_kernel<<<(nW1 + 255) / 256, 256, 0, stream>>>(mlp_w1, w1h);
        const int nHH = LAYERS * (int)HH;
        f32_to_f16_kernel<<<(nHH + 255) / 256, 256, 0, stream>>>(mlp_w2, w2h, nHH);
        f32_to_f16_kernel<<<(nHH + 255) / 256, 256, 0, stream>>>(lin1_w, l1h, nHH);
        f32_to_f16_kernel<<<(nHH + 255) / 256, 256, 0, stream>>>(lin2_w, l2h, nHH);
        f32_to_f16_kernel<<<(nHH + 255) / 256, 256, 0, stream>>>(lin_w, lwh, nHH);
        const int nPJ = (int)HH;
        f32_to_f16_kernel<<<(nPJ + 255) / 256, 256, 0, stream>>>(proj_w, pjh, nPJ);
    }

    // ---- edge features + embedding ----
    edge_geom_kernel<<<(NEDGES + 255) / 256, 256, 0, stream>>>(pos, ei, ea, Cbuf);
    embed_kernel<<<(NH + 255) / 256, 256, 0, stream>>>(z, emb, hbuf, NH);

    const dim3 blk(256);
    const dim3 grdE((NEDGES + 127) / 128, HDIM / 32);
    const dim3 grdN((NNODES + 127) / 128, HDIM / 32);

    // ---- interaction layers ----
    for (int l = 0; l < LAYERS; ++l) {
        const _Float16* w1l = w1h + (size_t)l * GPAD * HDIM;
        const _Float16* w2l = w2h + (size_t)l * HH;
        const _Float16* l1l = l1h + (size_t)l * HH;
        const _Float16* l2l = l2h + (size_t)l * HH;
        const _Float16* lwl = lwh + (size_t)l * HH;
        const float* b1l = mlp_b1 + (size_t)l * HDIM;
        const float* b2l = mlp_b2 + (size_t)l * HDIM;
        const float* lb2 = lin2_b + (size_t)l * HDIM;
        const float* lbl = lin_b  + (size_t)l * HDIM;

        // agg = 0
        fill_zero_kernel<<<(NH + 255) / 256, 256, 0, stream>>>(aggb, NH);

        // x1 = h @ lin1_w  (f32 A staged->f16, f16 out, no bias)
        wmma_gemm_kernel<EPI_F16, float><<<grdN, blk, 0, stream>>>(
            hbuf, HDIM, NNODES, l1l, HDIM, HDIM, nullptr,
            x1h, nullptr, nullptr, nullptr, nullptr, nullptr, nullptr);

        // t = ssp(edge_attr @ mlp_w1 + b1)
        wmma_gemm_kernel<EPI_SSP_F16, _Float16><<<grdE, blk, 0, stream>>>(
            ea, GPAD, NEDGES, w1l, HDIM, GPAD, b1l,
            tbuf, nullptr, nullptr, nullptr, nullptr, nullptr, nullptr);

        // Wf = (t @ mlp_w2 + b2) * C;  agg[dst] += x1[src] * Wf   (fused scatter)
        wmma_gemm_kernel<EPI_SCATTER, _Float16><<<grdE, blk, 0, stream>>>(
            tbuf, HDIM, NEDGES, w2l, HDIM, HDIM, b2l,
            nullptr, nullptr, Cbuf, ei, ei + NEDGES, x1h, aggb);

        // s = ssp(agg @ lin2_w + lin2_b)
        wmma_gemm_kernel<EPI_SSP_F16, float><<<grdN, blk, 0, stream>>>(
            aggb, HDIM, NNODES, l2l, HDIM, HDIM, lb2,
            sbuf, nullptr, nullptr, nullptr, nullptr, nullptr, nullptr);

        // h += s @ lin_w + lin_b
        wmma_gemm_kernel<EPI_RESID, _Float16><<<grdN, blk, 0, stream>>>(
            sbuf, HDIM, NNODES, lwl, HDIM, HDIM, lbl,
            nullptr, hbuf, nullptr, nullptr, nullptr, nullptr, nullptr);
    }

    // ---- projection: p = h @ proj_w + proj_b  (straight into d_out) ----
    float* pout = (float*)d_out;
    wmma_gemm_kernel<EPI_F32, float><<<grdN, blk, 0, stream>>>(
        hbuf, HDIM, NNODES, pjh, PDIM, HDIM, proj_b,
        nullptr, pout, nullptr, nullptr, nullptr, nullptr, nullptr);

    // ---- LayerNorm + SiLU in place on d_out ----
    ln_silu_kernel<<<NNODES, 256, 0, stream>>>(pout, ln_g, ln_b);

    // ---- second tuple output: batch (raw int bits appended after p) ----
    hipMemcpyAsync((char*)d_out + (size_t)NNODES * PDIM * sizeof(float),
                   batch, (size_t)NNODES * sizeof(int),
                   hipMemcpyDeviceToDevice, stream);
}
